// Glm4vMoeTextExperts_64020782514366
// MI455X (gfx1250) — compile-verified, matrix operand order checked by reading, not run on hardware
//
#include <hip/hip_runtime.h>
#include <math.h>

// ---------------------------------------------------------------------------
// MoE experts for MI455X (gfx1250, wave32, WMMA bf16 16x16x32)
//   E=16, T=2048, H=1024, I=512, top-k=2
// Pass 0: route tokens -> per-expert lists; convert X to bf16 once.
// Pass 1: fp32 weights -> bf16 TRANSPOSED copies in workspace (L2-resident,
//         48MB << 192MB L2): every WMMA B-fragment is one contiguous 32-byte
//         per-lane load, zero cvt work in the hot loop.
// Pass 2: fused gate_up GEMM -> silu-gate -> down GEMM -> weighted scatter.
//         M-tile = 32 rows/block (halves L2 re-reads, 2x B reuse); X tile
//         staged via global_load_async_to_lds_b128 (ASYNCcnt path); all B
//         fragments of a k-step staged in registers for deep load pipelining.
// ---------------------------------------------------------------------------

#define NE     16
#define NT     2048
#define NH     1024
#define NI     512
#define TOPK   2
#define CAP    (NT * TOPK)     // worst-case tokens per expert = 4096
#define MROWS  32              // rows (token,expert pairs) per block
#define MTILES (CAP / MROWS)   // 128 row-tiles per expert max

// padded LDS strides (bf16 elements); 32-byte aligned rows for v16bf loads
#define XSTR   (NH + 16)       // 1040 halfs = 2080 B (65*32)
#define ASTR   (NI + 16)       // 528  halfs = 1056 B (33*32)

typedef __attribute__((ext_vector_type(16))) __bf16 v16bf;
typedef __attribute__((ext_vector_type(8)))  __bf16 v8bf;
typedef __attribute__((ext_vector_type(8)))  float  v8f;

// ---------------------------------------------------------------------------
// Phase 0a: routing — build per-expert token lists in workspace
// ---------------------------------------------------------------------------
__global__ void route_kernel(const int* __restrict__ topk_idx,
                             const float* __restrict__ topk_w,
                             int* __restrict__ cnt,
                             int* __restrict__ pairTok,
                             float* __restrict__ pairW) {
    int t = blockIdx.x * blockDim.x + threadIdx.x;
    if (t >= NT) return;
#pragma unroll
    for (int k = 0; k < TOPK; ++k) {
        int e   = topk_idx[t * TOPK + k];
        float w = topk_w[t * TOPK + k];
        int slot = atomicAdd(&cnt[e], 1);
        pairTok[e * CAP + slot] = t;
        pairW[e * CAP + slot]   = w;
    }
}

// ---------------------------------------------------------------------------
// Phase 0b: hidden_states [T,H] fp32 -> bf16 (each token converted once)
// ---------------------------------------------------------------------------
__global__ void cvt_x(const float* __restrict__ in, __bf16* __restrict__ outb) {
    const size_t flat = ((size_t)blockIdx.x * blockDim.x + threadIdx.x) * 8;
    if (flat >= (size_t)NT * NH) return;
    const float4 v0 = *(const float4*)(in + flat);
    const float4 v1 = *(const float4*)(in + flat + 4);
    v8bf t;
    t[0] = (__bf16)v0.x; t[1] = (__bf16)v0.y; t[2] = (__bf16)v0.z; t[3] = (__bf16)v0.w;
    t[4] = (__bf16)v1.x; t[5] = (__bf16)v1.y; t[6] = (__bf16)v1.z; t[7] = (__bf16)v1.w;
    *(v8bf*)(outb + flat) = t;
}

// ---------------------------------------------------------------------------
// Phase 1a: gate_up_proj [E,H,2I] fp32  ->  WgT [E,2I,H] bf16 (transposed)
// ---------------------------------------------------------------------------
__global__ void cvt_transpose_gup(const float* __restrict__ in,
                                  __bf16* __restrict__ outT) {
    const size_t flat = ((size_t)blockIdx.x * blockDim.x + threadIdx.x) * 8;
    if (flat >= (size_t)NE * (2 * NI) * NH) return;
    const int e    = (int)(flat / ((size_t)(2 * NI) * NH));
    const int rem  = (int)(flat % ((size_t)(2 * NI) * NH));
    const int ncol = rem / NH;
    const int k0   = rem % NH;
    const float* __restrict__ src = in + ((size_t)e * NH + k0) * (2 * NI) + ncol;
    v8bf t;
#pragma unroll
    for (int j = 0; j < 8; ++j) t[j] = (__bf16)src[(size_t)j * (2 * NI)];
    *(v8bf*)(outT + flat) = t;
}

// ---------------------------------------------------------------------------
// Phase 1b: down_proj [E,I,H] fp32  ->  WdT [E,H,I] bf16 (transposed)
// ---------------------------------------------------------------------------
__global__ void cvt_transpose_down(const float* __restrict__ in,
                                   __bf16* __restrict__ outT) {
    const size_t flat = ((size_t)blockIdx.x * blockDim.x + threadIdx.x) * 8;
    if (flat >= (size_t)NE * NH * NI) return;
    const int e    = (int)(flat / ((size_t)NH * NI));
    const int rem  = (int)(flat % ((size_t)NH * NI));
    const int ncol = rem / NI;
    const int k0   = rem % NI;
    const float* __restrict__ src = in + ((size_t)e * NI + k0) * NH + ncol;
    v8bf t;
#pragma unroll
    for (int j = 0; j < 8; ++j) t[j] = (__bf16)src[(size_t)j * NH];
    *(v8bf*)(outT + flat) = t;
}

// ---------------------------------------------------------------------------
// Phase 2: fused expert compute. One block = (expert e, 32-row tile).
// 256 threads = 8 wave32s. Per-wave ownership:
//   Phase A: 64 gate cols + 64 up cols  (4 n-tiles x 2 m-subtiles -> 16 WMMA/k)
//   Phase B: 128 output cols            (8 n-tiles x 2 m-subtiles -> 16 WMMA/k)
// Each B fragment (one 32-byte per-lane load of transposed bf16 weights,
// lanes 0-15: K=kb..kb+15 of col n; lanes 16-31: K=kb+16..kb+31) feeds 2 WMMAs.
// All 8 B fragments of a k-step are staged in registers before the WMMA burst
// so the wave keeps many loads in flight (staged s_wait_loadcnt retirement).
// ---------------------------------------------------------------------------
__launch_bounds__(256)
__global__ void moe_expert_kernel(const __bf16* __restrict__ xbf,     // [T,H]
                                  const __bf16* __restrict__ WgT,     // [E,2I,H]
                                  const __bf16* __restrict__ WdT,     // [E,H,I]
                                  const int*    __restrict__ cnt,
                                  const int*    __restrict__ pairTok,
                                  const float*  __restrict__ pairW,
                                  float*        __restrict__ out) {   // [T,H]
    const int e  = blockIdx.y;
    const int n  = cnt[e];
    const int m0 = blockIdx.x * MROWS;
    if (m0 >= n) return;                       // uniform per block
    const int rows = min(MROWS, n - m0);

    __shared__ __bf16 ldsX[MROWS * XSTR];      // 32 x H activation tile (bf16)
    __shared__ __bf16 ldsA[MROWS * ASTR];      // 32 x I silu(g)*u tile (bf16)
    __shared__ int    sTok[MROWS];
    __shared__ float  sW[MROWS];

    const int tid   = threadIdx.x;
    const int wave  = tid >> 5;
    const int lane  = tid & 31;
    const int col   = lane & 15;               // N index inside a 16-wide tile
    const int khalf = (lane < 16) ? 0 : 16;    // K-half per A/B layout
    const int mhi   = (lane < 16) ? 0 : 8;     // C-layout row offset

    if (tid < MROWS) {
        const bool valid = tid < rows;
        sTok[tid] = valid ? pairTok[e * CAP + m0 + tid] : 0;
        sW[tid]   = valid ? pairW[e * CAP + m0 + tid]   : 0.0f;
    }
    __syncthreads();

    // --- stage X tile (32 x H bf16) via async global->LDS copies ----------
    {
        const uint32_t ldsX_base =
            (uint32_t)(size_t)(__attribute__((address_space(3))) __bf16*)&ldsX[0];
        const int CHUNKS = (NH * 2) / 16;      // 128 x 16-byte chunks per row
        for (int i = tid; i < MROWS * CHUNKS; i += 256) {
            const int r = i / CHUNKS;
            const int c = i % CHUNKS;
            if (r < rows) {
                const uint32_t ldst = ldsX_base + (uint32_t)(r * XSTR * 2 + c * 16);
                const unsigned long long ga =
                    (unsigned long long)(size_t)(xbf + (size_t)sTok[r] * NH + c * 8);
                asm volatile("global_load_async_to_lds_b128 %0, %1, off"
                             :: "v"(ldst), "v"(ga) : "memory");
            } else {
                v8bf z = {};
                *(v8bf*)&ldsX[r * XSTR + c * 8] = z;   // zero-pad invalid rows
            }
        }
        asm volatile("s_wait_asynccnt 0x0" ::: "memory");
    }
    __syncthreads();

    // =================== Phase A: gate_up GEMM + silu-gate ================
    const __bf16* __restrict__ Wg = WgT + (size_t)e * (2 * NI) * NH;
    const int nbase = wave * 64;               // this wave's gate-column base

    v8f accg[2][4] = {};
    v8f accu[2][4] = {};

    for (int kb = 0; kb < NH; kb += 32) {
        // A fragments: 32 contiguous LDS bytes, k = kb+khalf..+15 (same k order as B)
        const v16bf a0 = *(const v16bf*)&ldsX[(lane & 15) * XSTR + kb + khalf];
        const v16bf a1 = *(const v16bf*)&ldsX[(16 + (lane & 15)) * XSTR + kb + khalf];

        // stage all 8 B fragments (64 VGPRs) -> one big load clause in flight
        v16bf bg[4], bu[4];
#pragma unroll
        for (int nt = 0; nt < 4; ++nt) {
            const int ncol = nbase + nt * 16 + col;
            const __bf16* __restrict__ pg = Wg + (size_t)ncol * NH + kb + khalf;
            __builtin_prefetch(pg + 32, 0, 1);               // next k-step
            bg[nt] = *(const v16bf*)pg;                      // gate column frag
            bu[nt] = *(const v16bf*)(pg + (size_t)NI * NH);  // up column frag
        }
#pragma unroll
        for (int nt = 0; nt < 4; ++nt) {
            accg[0][nt] = __builtin_amdgcn_wmma_f32_16x16x32_bf16(
                false, a0, false, bg[nt], (short)0, accg[0][nt], false, false);
            accg[1][nt] = __builtin_amdgcn_wmma_f32_16x16x32_bf16(
                false, a1, false, bg[nt], (short)0, accg[1][nt], false, false);
            accu[0][nt] = __builtin_amdgcn_wmma_f32_16x16x32_bf16(
                false, a0, false, bu[nt], (short)0, accu[0][nt], false, false);
            accu[1][nt] = __builtin_amdgcn_wmma_f32_16x16x32_bf16(
                false, a1, false, bu[nt], (short)0, accu[1][nt], false, false);
        }
    }

    // silu(g) * u -> LDS activation tile (C layout: VGPR r holds row r / r+8)
#pragma unroll
    for (int ms = 0; ms < 2; ++ms) {
#pragma unroll
        for (int nt = 0; nt < 4; ++nt) {
#pragma unroll
            for (int r = 0; r < 8; ++r) {
                const int m = ms * 16 + r + mhi;
                const float g = accg[ms][nt][r];
                const float u = accu[ms][nt][r];
                const float s = g / (1.0f + __expf(-g));
                ldsA[m * ASTR + nbase + nt * 16 + col] = (__bf16)(s * u);
            }
        }
    }
    __syncthreads();

    // =================== Phase B: down GEMM + scaled scatter ==============
    const __bf16* __restrict__ Wd = WdT + (size_t)e * NH * NI;
    const int ob = wave * 128;                 // this wave's output-column base

    v8f acc[2][8] = {};

    for (int kb = 0; kb < NI; kb += 32) {
        const v16bf a0 = *(const v16bf*)&ldsA[(lane & 15) * ASTR + kb + khalf];
        const v16bf a1 = *(const v16bf*)&ldsA[(16 + (lane & 15)) * ASTR + kb + khalf];

        v16bf b[8];
#pragma unroll
        for (int nt = 0; nt < 8; ++nt) {
            const int ncol = ob + nt * 16 + col;
            const __bf16* __restrict__ pd = Wd + (size_t)ncol * NI + kb + khalf;
            __builtin_prefetch(pd + 32, 0, 1);
            b[nt] = *(const v16bf*)pd;
        }
#pragma unroll
        for (int nt = 0; nt < 8; ++nt) {
            acc[0][nt] = __builtin_amdgcn_wmma_f32_16x16x32_bf16(
                false, a0, false, b[nt], (short)0, acc[0][nt], false, false);
            acc[1][nt] = __builtin_amdgcn_wmma_f32_16x16x32_bf16(
                false, a1, false, b[nt], (short)0, acc[1][nt], false, false);
        }
    }

    // scale by routing weight, accumulate into out (duplicates accumulate)
#pragma unroll
    for (int ms = 0; ms < 2; ++ms) {
#pragma unroll
        for (int nt = 0; nt < 8; ++nt) {
#pragma unroll
            for (int r = 0; r < 8; ++r) {
                const int m = ms * 16 + r + mhi;
                if (m < rows) {
                    atomicAdd(&out[(size_t)sTok[m] * NH + ob + nt * 16 + col],
                              acc[ms][nt][r] * sW[m]);
                }
            }
        }
    }
}

// ---------------------------------------------------------------------------
// Host-side launcher (graph-capture safe: only async ops on `stream`)
// ---------------------------------------------------------------------------
extern "C" void kernel_launch(void* const* d_in, const int* in_sizes, int n_in,
                              void* d_out, int out_size, void* d_ws, size_t ws_size,
                              hipStream_t stream) {
    const float* x    = (const float*)d_in[0];   // hidden_states [T,H]
    const float* tw   = (const float*)d_in[1];   // topk_weights  [T,K]
    const int*   ti   = (const int*)d_in[2];     // topk_indices  [T,K]
    const float* gup  = (const float*)d_in[3];   // gate_up_proj  [E,H,2I]
    const float* down = (const float*)d_in[4];   // down_proj     [E,I,H]
    float*       out  = (float*)d_out;           // [T,H]

    // workspace layout (all offsets 256B-aligned):
    //   cnt[16]            @ 0
    //   pairTok[E*CAP]     @ 256            (256 KB)
    //   pairW[E*CAP]       @ ~256 KB        (256 KB)
    //   WgT bf16 [E,2I,H]  @ 1 MB           (32 MB)
    //   WdT bf16 [E,H,I]   @ 33 MB          (16 MB)
    //   Xbf bf16 [T,H]     @ 49 MB          (4 MB)
    char* ws = (char*)d_ws;
    int*    cnt     = (int*)ws;
    int*    pairTok = (int*)(ws + 256);
    float*  pairW   = (float*)(ws + 256 + (size_t)NE * CAP * sizeof(int));
    __bf16* WgT     = (__bf16*)(ws + (1ull << 20));
    __bf16* WdT     = (__bf16*)(ws + (1ull << 20) +
                                (size_t)NE * (2 * NI) * NH * sizeof(__bf16));
    __bf16* Xbf     = (__bf16*)(ws + (1ull << 20) +
                                ((size_t)NE * (2 * NI) * NH +
                                 (size_t)NE * NH * NI) * sizeof(__bf16));

    hipMemsetAsync(cnt, 0, 64, stream);
    hipMemsetAsync(d_out, 0, (size_t)NT * NH * sizeof(float), stream);

    route_kernel<<<(NT + 255) / 256, 256, 0, stream>>>(ti, tw, cnt, pairTok, pairW);

    {   // one-off conversion passes (bf16 copies stay L2-resident)
        const size_t nx = (size_t)NT * NH / 8;
        const size_t ng = (size_t)NE * (2 * NI) * NH / 8;
        const size_t nd = (size_t)NE * NH * NI / 8;
        cvt_x<<<(unsigned)((nx + 255) / 256), 256, 0, stream>>>(x, Xbf);
        cvt_transpose_gup<<<(unsigned)((ng + 255) / 256), 256, 0, stream>>>(gup, WgT);
        cvt_transpose_down<<<(unsigned)((nd + 255) / 256), 256, 0, stream>>>(down, WdT);
    }

    dim3 grid(MTILES, NE);
    moe_expert_kernel<<<grid, 256, 0, stream>>>(Xbf, WgT, WdT, cnt, pairTok, pairW, out);
}
